// HybridMatrixNeuralODE_31988916420753
// MI455X (gfx1250) — matrix operand order repaired and checked by reading, not compiled, not run on hardware
//
#include <hip/hip_runtime.h>
#include <hip/hip_bf16.h>

// ---------------------------------------------------------------------------
// out = tanh(x @ (M@W1) + b1) @ W2 + b2,  M = (I + 0.02*A)^50
// Phase 1 (1 block, f32): M via repeated squaring (7 matmuls), W=M@W1 -> f16
//   W^T, and W2 -> f16 W2^T, into d_ws.
// Phase 2 (4096 blocks): fused GEMM1+tanh+GEMM2 with v_wmma_f32_16x16x32_f16.
// ---------------------------------------------------------------------------

typedef _Float16 v16h __attribute__((ext_vector_type(16)));
typedef _Float16 v8h  __attribute__((ext_vector_type(8)));
typedef float    v8f  __attribute__((ext_vector_type(8)));
typedef float    v4f  __attribute__((ext_vector_type(4)));

union U16 { v16h v; v8h h2[2]; _Float16 h[16]; };

__device__ __forceinline__ v8f wmma_f16(v16h a, v16h b, v8f c) {
  // D = A(16x32 f16) * B(32x16 f16) + C(16x16 f32)
  return __builtin_amdgcn_wmma_f32_16x16x32_f16(false, a, false, b,
                                                (short)0, c, false, false);
}

__device__ __forceinline__ float fast_tanh(float x) {
#if __has_builtin(__builtin_amdgcn_tanhf)
  return __builtin_amdgcn_tanhf(x);
#else
  float r;
  // gfx1250 transcendental; v_nop satisfies the 1-op TRANS hazard gap.
  asm volatile("v_tanh_f32 %0, %1\n\tv_nop" : "=v"(r) : "v"(x));
  return r;
#endif
}

// ---------------------------------------------------------------------------
// Phase 1: single-block prep (f32, exact w.r.t. the reference chain)
// ---------------------------------------------------------------------------
__device__ __forceinline__ void mm64(float* dst, const float* s1,
                                     const float* s2, int tid) {
  // 64x64 = 64x64 * 64x64 ; one 4x4 tile per thread (256 threads)
  const int tr = (tid >> 4) * 4;
  const int tc = (tid & 15) * 4;
  float acc[4][4] = {};
  for (int k = 0; k < 64; ++k) {
    float a0 = s1[(tr + 0) * 64 + k];
    float a1 = s1[(tr + 1) * 64 + k];
    float a2 = s1[(tr + 2) * 64 + k];
    float a3 = s1[(tr + 3) * 64 + k];
    float b0 = s2[k * 64 + tc + 0];
    float b1 = s2[k * 64 + tc + 1];
    float b2 = s2[k * 64 + tc + 2];
    float b3 = s2[k * 64 + tc + 3];
    acc[0][0] += a0 * b0; acc[0][1] += a0 * b1; acc[0][2] += a0 * b2; acc[0][3] += a0 * b3;
    acc[1][0] += a1 * b0; acc[1][1] += a1 * b1; acc[1][2] += a1 * b2; acc[1][3] += a1 * b3;
    acc[2][0] += a2 * b0; acc[2][1] += a2 * b1; acc[2][2] += a2 * b2; acc[2][3] += a2 * b3;
    acc[3][0] += a3 * b0; acc[3][1] += a3 * b1; acc[3][2] += a3 * b2; acc[3][3] += a3 * b3;
  }
  #pragma unroll
  for (int i = 0; i < 4; ++i)
    #pragma unroll
    for (int j = 0; j < 4; ++j)
      dst[(tr + i) * 64 + (tc + j)] = acc[i][j];
  __syncthreads();
}

__global__ __launch_bounds__(256) void prep_kernel(
    const float* __restrict__ Ag, const float* __restrict__ W1g,
    const float* __restrict__ W2g, _Float16* __restrict__ Wtg,
    _Float16* __restrict__ W2tg) {
  __shared__ alignas(32) float sB[64 * 64];  // B = I + h*A
  __shared__ alignas(32) float sX[64 * 64];
  __shared__ alignas(32) float sY[64 * 64];
  const int tid = threadIdx.x;

  for (int i = tid; i < 4096; i += 256) {
    int r = i >> 6, c = i & 63;
    float v = 0.02f * Ag[i];
    if (r == c) v += 1.0f;
    sB[i] = v;
    sX[i] = v;  // cur = B^1
  }
  __syncthreads();

  // 50 = 110010b : square-and-multiply, MSB first -> 5 squarings + 2 mults
  float* cur = sX;
  float* tmp = sY;
  const int bits[5] = {1, 0, 0, 1, 0};
  for (int it = 0; it < 5; ++it) {
    mm64(tmp, cur, cur, tid);                 // cur^2
    { float* t = cur; cur = tmp; tmp = t; }
    if (bits[it]) {
      mm64(tmp, cur, sB, tid);                // * B
      { float* t = cur; cur = tmp; tmp = t; }
    }
  }
  // cur == M = (I + hA)^50

  // Wt[n*64 + d] = (f16)( sum_k M[d][k] * W1[k][n] )   (N-major, K contiguous)
  for (int o = tid; o < 64 * 256; o += 256) {
    int d = o >> 8;
    int n = o & 255;
    float acc = 0.f;
    for (int k = 0; k < 64; ++k) acc += cur[d * 64 + k] * W1g[k * 256 + n];
    Wtg[n * 64 + d] = (_Float16)acc;
  }
  // W2t[n*256 + k] = (f16) W2[k][n]
  for (int o = tid; o < 16 * 256; o += 256) {
    int n = o >> 8, k = o & 255;
    W2tg[n * 256 + k] = (_Float16)W2g[k * 16 + n];
  }
}

// ---------------------------------------------------------------------------
// Phase 2: fused  out = tanh(x@W + b1) @ W2 + b2
// 256 threads = 8 waves; each wave: 2 strips of 16 rows (256 rows / block).
// ---------------------------------------------------------------------------
__global__ __launch_bounds__(256) void node_head_kernel(
    const float* __restrict__ xg, const float* __restrict__ b1g,
    const float* __restrict__ b2g, const _Float16* __restrict__ Wtg,
    const _Float16* __restrict__ W2tg, float* __restrict__ outg) {
  __shared__ alignas(32) _Float16 sWt[64 * 256];        // [n][k], 32 KB
  __shared__ alignas(32) _Float16 sW2t[16 * 256];       // [n][k],  8 KB
  __shared__ alignas(32) _Float16 sH[8 * 2 * 16 * 32];  // per-wave h staging, 16 KB

  const int tid = threadIdx.x;
  {  // cooperative fill of B-operand tables (hot for whole block)
    const uint4* s0 = (const uint4*)Wtg;
    uint4* d0 = (uint4*)sWt;
    for (int i = tid; i < (64 * 256) / 8; i += 256) d0[i] = s0[i];
    const uint4* s1 = (const uint4*)W2tg;
    uint4* d1 = (uint4*)sW2t;
    for (int i = tid; i < (16 * 256) / 8; i += 256) d1[i] = s1[i];
  }
  __syncthreads();

  const int lane = tid & 31;
  const int wave = tid >> 5;
  const int col = lane & 15;          // N (B/C/D) or M (A) index for this lane
  const int mhi = (lane >> 4) & 1;    // lane group 0..15 vs 16..31
  const size_t rowbase = (size_t)blockIdx.x * 256 + (size_t)wave * 32;

  // --- A operands for GEMM1: x rows, f32 -> f16, direct to A-layout regs ---
  // A 16x32 f16, k order used here (same as B): lanes<16 hold K {0..15}, lanes>=16 hold K {16..31}
  U16 ax[2][2];
  const int koff = mhi ? 16 : 0;
  #pragma unroll
  for (int s = 0; s < 2; ++s) {
    const float* xr = xg + (rowbase + (size_t)s * 16 + col) * 64;
    #pragma unroll
    for (int c = 0; c < 2; ++c) {
      #pragma unroll
      for (int g = 0; g < 2; ++g) {
        const int kb = c * 32 + g * 8 + koff;
        v4f q0 = *(const v4f*)(xr + kb);
        v4f q1 = *(const v4f*)(xr + kb + 4);
        #pragma unroll
        for (int j = 0; j < 4; ++j) {
          ax[s][c].h[g * 8 + j] = (_Float16)q0[j];
          ax[s][c].h[g * 8 + 4 + j] = (_Float16)q1[j];
        }
      }
    }
  }

  _Float16* hsBase = sH + wave * (2 * 512);  // two 16x32 f16 tiles per wave
  const int kbase = mhi ? 16 : 0;            // B-operand K group for this lane
  v8f oacc0 = {};
  v8f oacc1 = {};

  v8f Cst[2];  // C tiles of the current n (strip 0 / strip 1)

  auto gemm1_tile = [&](int n) {
    U16 bw0, bw1;  // W tiles (B operand): K contiguous per lane, N = col
    bw0.v = *(const v16h*)&sWt[(n * 16 + col) * 64 + 0 * 32 + kbase];
    bw1.v = *(const v16h*)&sWt[(n * 16 + col) * 64 + 1 * 32 + kbase];
    v8f C0 = {}, C1 = {};
    C0 = wmma_f16(ax[0][0].v, bw0.v, C0);
    C0 = wmma_f16(ax[0][1].v, bw1.v, C0);
    C1 = wmma_f16(ax[1][0].v, bw0.v, C1);
    C1 = wmma_f16(ax[1][1].v, bw1.v, C1);
    const float b1v = b1g[n * 16 + col];
    Cst[0] = C0;
    Cst[1] = C1;
    #pragma unroll
    for (int s = 0; s < 2; ++s) {
      // C layout: VGPR r, lanes<16 -> row r, lanes>=16 -> row r+8; N = col.
      _Float16* hs = hsBase + s * 512 + (n & 1) * 16 + col;
      #pragma unroll
      for (int r = 0; r < 8; ++r) {
        float t = fast_tanh(Cst[s][r] + b1v);
        hs[(mhi * 8 + r) * 32] = (_Float16)t;  // sH[m][kk], row stride 32 halves
      }
    }
  };

  for (int q = 0; q < 8; ++q) {
    gemm1_tile(2 * q);
    gemm1_tile(2 * q + 1);
    // --- GEMM2 accumulate: out += h_chunk(16x32) @ W2_chunk(32x16) ---
    U16 a20, a21, b2w;
    const int kg = mhi ? 16 : 0;
    const _Float16* p0 = hsBase + 0 * 512 + col * 32;
    a20.h2[0] = *(const v8h*)(p0 + kg);
    a20.h2[1] = *(const v8h*)(p0 + 8 + kg);
    const _Float16* p1 = hsBase + 1 * 512 + col * 32;
    a21.h2[0] = *(const v8h*)(p1 + kg);
    a21.h2[1] = *(const v8h*)(p1 + 8 + kg);
    b2w.v = *(const v16h*)&sW2t[col * 256 + q * 32 + kbase];
    oacc0 = wmma_f16(a20.v, b2w.v, oacc0);
    oacc1 = wmma_f16(a21.v, b2w.v, oacc1);
  }

  // --- epilogue: + b2, store 16x16 f32 per strip ---
  const float bb = b2g[col];
  #pragma unroll
  for (int s = 0; s < 2; ++s) {
    const v8f O = s ? oacc1 : oacc0;
    float* orow = outg + (rowbase + (size_t)s * 16) * 16;
    #pragma unroll
    for (int r = 0; r < 8; ++r)
      orow[(mhi * 8 + r) * 16 + col] = O[r] + bb;
  }
}

// ---------------------------------------------------------------------------
extern "C" void kernel_launch(void* const* d_in, const int* in_sizes, int n_in,
                              void* d_out, int out_size, void* d_ws,
                              size_t ws_size, hipStream_t stream) {
  const float* x  = (const float*)d_in[0];   // (B, 64)
  const float* A  = (const float*)d_in[1];   // (64, 64)
  const float* W1 = (const float*)d_in[2];   // (64, 256)
  const float* b1 = (const float*)d_in[3];   // (256,)
  const float* W2 = (const float*)d_in[4];   // (256, 16)
  const float* b2 = (const float*)d_in[5];   // (16,)
  float* out = (float*)d_out;                // (B, 16) f32

  _Float16* Wt  = (_Float16*)d_ws;           // 64*256 f16 (W^T, N-major)
  _Float16* W2t = Wt + 64 * 256;             // 16*256 f16 (W2^T, N-major)

  prep_kernel<<<1, 256, 0, stream>>>(A, W1, W2, Wt, W2t);

  const int rows = in_sizes[0] / 64;         // 1048576; multiple of 256
  node_head_kernel<<<dim3(rows / 256), 256, 0, stream>>>(x, b1, b2, Wt, W2t,
                                                         out);
}
